// SpatialAttnProcessor2_0_42537356100134
// MI455X (gfx1250) — matrix operand order repaired and checked by reading, not compiled, not run on hardware
//
#include <hip/hip_runtime.h>

// ---------------------------------------------------------------------------
// CDNA5 / gfx1250 bf16-WMMA implementation of the 5-image cross-attention
// block.  Matmuls on v_wmma_f32_16x16x32_bf16 (wave32); gathered K staging
// via global_load_async_to_lds_b128 (ASYNCcnt DMA); GEMM A-tiles streamed by
// the Tensor Data Mover (tensor_load_to_lds, TENSORcnt) with double buffering.
// ---------------------------------------------------------------------------

typedef __attribute__((ext_vector_type(16))) __bf16 v16bf;
typedef __attribute__((ext_vector_type(8)))  __bf16 v8bf;
typedef __attribute__((ext_vector_type(8)))  float  v8f;
typedef __attribute__((ext_vector_type(4)))  unsigned u32x4;
typedef __attribute__((ext_vector_type(8)))  int      i32x8;
typedef __attribute__((ext_vector_type(4)))  int      i32x4;

#define C_DIM   1280
#define T_DIM   1024
#define HEADS   20
#define HD      64
#define IMGS    5
#define NSEL    512
#define KV_LEN  3072            // 4*512 gathered + 1024 self
#define M_ROWS  10240           // 10 * 1024 total rows
#define KSTEPS  (C_DIM / 32)    // 40 k-chunks of 32 for WMMA
#define QW      4               // waves per attention block

#ifndef __has_builtin
#define __has_builtin(x) 0
#endif
#if __has_builtin(__builtin_amdgcn_tensor_load_to_lds)
#define HAVE_TDM 1
#else
#define HAVE_TDM 0
#endif

// ---------------------------------------------------------------------------
// fp32 -> bf16 elementwise convert (8 elements / thread)
// ---------------------------------------------------------------------------
__global__ void cvt_f32_to_bf16(const float* __restrict__ src,
                                __bf16* __restrict__ dst, int n8) {
  int t = blockIdx.x * blockDim.x + threadIdx.x;
  if (t >= n8) return;
  const float4* s = (const float4*)src + (size_t)t * 2;
  float4 a = s[0], b = s[1];
  v8bf o;
  o[0] = (__bf16)a.x; o[1] = (__bf16)a.y; o[2] = (__bf16)a.z; o[3] = (__bf16)a.w;
  o[4] = (__bf16)b.x; o[5] = (__bf16)b.y; o[6] = (__bf16)b.z; o[7] = (__bf16)b.w;
  *((v8bf*)dst + t) = o;
}

// ---------------------------------------------------------------------------
// Repack a 1280x1280 f32 weight (row-major KxN) into per-lane bf16
// B-fragments: for tile (nt, kt) lane l holds 16 contiguous bf16 with
// n = nt*16 + (l%16), K = kt*32 + (l/16)*8 + j (j<8), +16 for j>=8.   (CDNA5 ISA B layout)
// ---------------------------------------------------------------------------
__global__ void pack_w_bf16(const float* __restrict__ W, __bf16* __restrict__ dst) {
  int t = blockIdx.x * blockDim.x + threadIdx.x;   // (C/16)*KSTEPS*32 = 102400
  if (t >= (C_DIM / 16) * KSTEPS * 32) return;
  int lane = t & 31;
  int kt   = (t >> 5) % KSTEPS;
  int nt   = t / (32 * KSTEPS);
  int n    = nt * 16 + (lane & 15);
  int kb   = kt * 32 + (lane >> 4) * 8;
  v8bf lo, hi;
#pragma unroll
  for (int j = 0; j < 8; ++j) {
    lo[j] = (__bf16)W[(size_t)(kb + j) * C_DIM + n];
    hi[j] = (__bf16)W[(size_t)(kb + 16 + j) * C_DIM + n];
  }
  v8bf* d = (v8bf*)(dst + (size_t)t * 16);
  d[0] = lo; d[1] = hi;
}

// ---------------------------------------------------------------------------
// Build kv row-index table: kvrow[(i*2+r)*3072 + p] = global row into K/V.
// ---------------------------------------------------------------------------
__global__ void build_kvrow(const int* __restrict__ indices, int* __restrict__ kvrow) {
  int t = blockIdx.x * blockDim.x + threadIdx.x;   // 5*2*3072 = 30720
  if (t >= IMGS * 2 * KV_LEN) return;
  int i = t / (2 * KV_LEN);
  int r = (t / KV_LEN) & 1;
  int p = t % KV_LEN;
  int row;
  if (p < 4 * NSEL) {
    int jj = p / NSEL;
    int j  = jj + (jj >= i ? 1 : 0);
    int tt = indices[j * NSEL + (p % NSEL)];
    row = r * (IMGS * T_DIM) + j * T_DIM + tt;
  } else {
    row = r * (IMGS * T_DIM) + i * T_DIM + (p - 4 * NSEL);
  }
  kvrow[t] = row;
}

// ---------------------------------------------------------------------------
// WMMA GEMM: D = A(MxK bf16, row-major) * Bpacked.  Block = 4 waves,
// 128x64 tile; each wave 32x64 (two M-subtiles reuse each B fragment).
// A tiles (128x32) are streamed into a double-buffered LDS tile by the
// Tensor Data Mover (issued by wave 0, TENSORcnt-tracked) so the DMA of
// tile kt+1 overlaps compute on tile kt.  Falls back to direct VGPR loads
// when the TDM builtin is unavailable.
// MODE 0: bf16 out.  MODE 1: f32 out with bias + residual fused.
// ---------------------------------------------------------------------------
template <int MODE>
__global__ __launch_bounds__(128)
void gemm_wmma(const __bf16* __restrict__ A, const __bf16* __restrict__ Bpk,
               __bf16* __restrict__ outBf, float* __restrict__ outF,
               const float* __restrict__ bias, const float* __restrict__ res) {
  int lane = threadIdx.x & 31;
  int w    = threadIdx.x >> 5;
  int l16  = lane & 15;
  int lh   = lane >> 4;
  int mbase = blockIdx.y * 128 + w * 32;
  int nbase = blockIdx.x * 64;

#if HAVE_TDM
  __shared__ __align__(16) __bf16 ldsA[2][128 * 32];   // double-buffered A tile
  // TDM descriptor (CDNA5 ISA): 2D tensor, 2-byte
  // elements, tile 32 x 128, row stride C_DIM.  Uniform -> SGPRs.
  auto tdmIssueA = [&](int buf, int kt) {
    unsigned long long ga = (unsigned long long)(const void*)(
        A + (size_t)blockIdx.y * 128 * C_DIM + kt * 32);
    unsigned ldsAddr = (unsigned)(unsigned long long)(const void*)&ldsA[buf][0];
    u32x4 g0;
    g0[0] = 1u;                                   // count=1, user descriptor
    g0[1] = ldsAddr;                              // lds_addr (bytes)
    g0[2] = (unsigned)ga;                         // global_addr[31:0]
    g0[3] = (unsigned)((ga >> 32) & 0x1FFFFFFull) // global_addr[56:32]
            | 0x80000000u;                        // type=2 ("image")
    i32x8 g1;
    g1[0] = 0x00010000;                           // wg_mask=0, data_size=2B
    g1[1] = (int)(1280u << 16);                   // tensor_dim0 = C_DIM (lo16)
    g1[2] = (int)(10240u << 16);                  // dim0 hi=0 | tensor_dim1 lo
    g1[3] = (int)(32u << 16);                     // dim1 hi=0 | tile_dim0 = 32
    g1[4] = 128;                                  // tile_dim1 = 128, dim2 = 0
    g1[5] = 1280;                                 // tensor_dim0_stride lo32
    g1[6] = 0;                                    // stride0 hi | stride1 lo
    g1[7] = 0;
    i32x4 gz4 = {0, 0, 0, 0};                     // 2D: groups 2/3 unused
    i32x8 gz8 = {0, 0, 0, 0, 0, 0, 0, 0};        // clang-23 6-arg form
    __builtin_amdgcn_tensor_load_to_lds(g0, g1, gz4, gz4, gz8, 0);
  };
  if (w == 0) tdmIssueA(0, 0);
#endif

  v8f acc[2][4] = {};
  for (int kt = 0; kt < KSTEPS; ++kt) {
    int kb = kt * 32;
    v16bf af[2];
#if HAVE_TDM
    int buf = kt & 1;
    if (w == 0) __builtin_amdgcn_s_wait_tensorcnt((short)0);
    __syncthreads();                 // tile kt in LDS, visible to all waves
    if (w == 0 && kt + 1 < KSTEPS) tdmIssueA(buf ^ 1, kt + 1);
#pragma unroll
    for (int ms = 0; ms < 2; ++ms) {
      const __bf16* ar = &ldsA[buf][(w * 32 + ms * 16 + l16) * 32 + lh * 8];
      v8bf a0 = *(const v8bf*)ar;
      v8bf a1 = *(const v8bf*)(ar + 16);
#pragma unroll
      for (int j = 0; j < 8; ++j) { af[ms][j] = a0[j]; af[ms][j + 8] = a1[j]; }
    }
#else
#pragma unroll
    for (int ms = 0; ms < 2; ++ms) {
      const __bf16* arow =
          A + (size_t)(mbase + ms * 16 + l16) * C_DIM + kb + lh * 8;
      v8bf a0 = *(const v8bf*)arow;
      v8bf a1 = *(const v8bf*)(arow + 16);
#pragma unroll
      for (int j = 0; j < 8; ++j) { af[ms][j] = a0[j]; af[ms][j + 8] = a1[j]; }
    }
#endif
#pragma unroll
    for (int nt = 0; nt < 4; ++nt) {
      const __bf16* bp =
          Bpk + ((size_t)((nbase >> 4) + nt) * KSTEPS + kt) * 512 + lane * 16;
      v8bf b0 = *(const v8bf*)bp;
      v8bf b1 = *(const v8bf*)(bp + 8);
      v16bf bf;
#pragma unroll
      for (int j = 0; j < 8; ++j) { bf[j] = b0[j]; bf[j + 8] = b1[j]; }
      acc[0][nt] = __builtin_amdgcn_wmma_f32_16x16x32_bf16(
          false, af[0], false, bf, (short)0, acc[0][nt], false, false);
      acc[1][nt] = __builtin_amdgcn_wmma_f32_16x16x32_bf16(
          false, af[1], false, bf, (short)0, acc[1][nt], false, false);
    }
  }
  // Epilogue.  C/D layout: VGPR r -> row r + 8*lh, lane%16 -> column.
#pragma unroll
  for (int ms = 0; ms < 2; ++ms) {
#pragma unroll
    for (int nt = 0; nt < 4; ++nt) {
      int col = nbase + nt * 16 + l16;
#pragma unroll
      for (int r = 0; r < 8; ++r) {
        int m = mbase + ms * 16 + r + lh * 8;
        float v = acc[ms][nt][r];
        if (MODE == 0) {
          outBf[(size_t)m * C_DIM + col] = (__bf16)v;
        } else {
          outF[(size_t)m * C_DIM + col] =
              v + bias[col] + res[(size_t)m * C_DIM + col];
        }
      }
    }
  }
}

// ---------------------------------------------------------------------------
// Flash attention: block = 4 waves sharing one (i, rep, head); each wave owns
// one 16-row q tile.  K chunk (32 kv x 64 hd) DMA'd into LDS row-major with
// global_load_async_to_lds_b128 (per-lane gathered global addr + LDS addr,
// ASYNCcnt).  V chunk staged transposed ([hd][kv]).  Online softmax: row max
// via wave32 shuffles (masks 1..8 keep the C/D 16-lane halves independent);
// the row-sum recurrence runs on the matrix pipe as P @ ones (one extra
// WMMA/chunk) instead of 32 ds_bpermutes.
// ---------------------------------------------------------------------------
__global__ __launch_bounds__(128)
void attn_wmma(const __bf16* __restrict__ Q, const __bf16* __restrict__ K,
               const __bf16* __restrict__ V, const int* __restrict__ kvrow,
               __bf16* __restrict__ O) {
  __shared__ __align__(16) __bf16 ldsK[32 * HD];        // [kv][hd] row-major
  __shared__ __align__(16) __bf16 ldsVt[HD * 32];       // [hd][kv] transposed
  __shared__ __align__(16) __bf16 ldsP[QW][16 * 32];    // per-wave probs

  int tid  = threadIdx.x;
  int lane = tid & 31;
  int w    = tid >> 5;
  int l16  = lane & 15;
  int lh   = lane >> 4;
  int h    = blockIdx.y;
  int ir   = blockIdx.z;            // i*2 + rep
  int i    = ir >> 1;
  int rr   = ir & 1;
  const int* rows = kvrow + (size_t)ir * KV_LEN;
  int qt    = blockIdx.x * QW + w;
  int qbase = rr * (IMGS * T_DIM) + i * T_DIM + qt * 16;
  const float scale = 0.125f;       // 1/sqrt(64)

  // Staging split: 128 threads cover 32 kv rows x 4 segments of 16 hd elems.
  int srow = tid >> 2;              // kv row within chunk (0..31)
  int seg  = tid & 3;               // 16-element hd segment (0..3)

  // Resident Q A-fragments (two K=32 windows of hd=64)
  v16bf qf[2];
  {
    const __bf16* qrow = Q + (size_t)(qbase + l16) * C_DIM + h * HD;
#pragma unroll
    for (int kk = 0; kk < 2; ++kk) {
      v8bf a0 = *(const v8bf*)(qrow + kk * 32 + lh * 16);
      v8bf a1 = *(const v8bf*)(qrow + kk * 32 + lh * 16 + 8);
#pragma unroll
      for (int j = 0; j < 8; ++j) { qf[kk][j] = a0[j]; qf[kk][j + 8] = a1[j]; }
    }
  }

  // Constant all-ones B-fragment: P @ ones == per-row softmax partial sums.
  v16bf onesf;
#pragma unroll
  for (int j = 0; j < 16; ++j) onesf[j] = (__bf16)1.0f;

  v8f o[4] = {};
  v8f osum = {};                    // running denominator (every col equal)
  float mrun[8];
#pragma unroll
  for (int r = 0; r < 8; ++r) mrun[r] = -1e30f;

  for (int c = 0; c < KV_LEN; c += 32) {
    int grow = rows[c + srow];
    // --- K chunk: async DMA gathered rows -> LDS row-major (32B/thread) ---
    {
      unsigned long long ga =
          (unsigned long long)(K + (size_t)grow * C_DIM + h * HD + seg * 16);
      unsigned ldsOff =
          (unsigned)(unsigned long long)&ldsK[srow * HD + seg * 16];
      asm volatile("global_load_async_to_lds_b128 %0, %1, off"
                   :: "v"(ldsOff), "v"(ga) : "memory");
      asm volatile("global_load_async_to_lds_b128 %0, %1, off offset:16"
                   :: "v"(ldsOff), "v"(ga) : "memory");
    }
    // --- V chunk: load gathered row segment, store transposed in LDS ---
    {
      const v8bf* gv =
          (const v8bf*)(V + (size_t)grow * C_DIM + h * HD + seg * 16);
      v8bf c0 = gv[0], c1 = gv[1];
#pragma unroll
      for (int j = 0; j < 8; ++j) {
        ldsVt[(seg * 16 + j) * 32 + srow]     = c0[j];
        ldsVt[(seg * 16 + 8 + j) * 32 + srow] = c1[j];
      }
    }
    asm volatile("s_wait_asynccnt 0x0" ::: "memory");
    __syncthreads();   // staged K (async) + Vt (ds) visible to all waves

    // --- scores S = Q K^T, K B-fragments from shared LDS tile ---
    v8f s[2];
#pragma unroll
    for (int sb = 0; sb < 2; ++sb) {
      const __bf16* kr = ldsK + (sb * 16 + l16) * HD;
      v16bf b0, b1;
      v8bf t0 = *(const v8bf*)(kr + lh * 16);
      v8bf t1 = *(const v8bf*)(kr + lh * 16 + 8);
#pragma unroll
      for (int j = 0; j < 8; ++j) { b0[j] = t0[j]; b0[j + 8] = t1[j]; }
      t0 = *(const v8bf*)(kr + 32 + lh * 16);
      t1 = *(const v8bf*)(kr + 32 + lh * 16 + 8);
#pragma unroll
      for (int j = 0; j < 8; ++j) { b1[j] = t0[j]; b1[j + 8] = t1[j]; }
      v8f a = {};
      a = __builtin_amdgcn_wmma_f32_16x16x32_bf16(false, qf[0], false, b0,
                                                  (short)0, a, false, false);
      a = __builtin_amdgcn_wmma_f32_16x16x32_bf16(false, qf[1], false, b1,
                                                  (short)0, a, false, false);
      s[sb] = a;
    }

    // --- online softmax (per accumulator row r = qrow r + 8*lh) ---
#pragma unroll
    for (int r = 0; r < 8; ++r) {
      float s0 = s[0][r] * scale, s1 = s[1][r] * scale;
      float mx = fmaxf(s0, s1);
#pragma unroll
      for (int msk = 1; msk < 16; msk <<= 1)
        mx = fmaxf(mx, __shfl_xor(mx, msk, 32));
      float nm    = fmaxf(mrun[r], mx);
      float alpha = __expf(mrun[r] - nm);
      float p0 = __expf(s0 - nm), p1 = __expf(s1 - nm);
      mrun[r] = nm;
      osum[r] *= alpha;
#pragma unroll
      for (int nt = 0; nt < 4; ++nt) o[nt][r] *= alpha;
      int m = r + lh * 8;
      ldsP[w][m * 32 + l16]      = (__bf16)p0;   // per-wave: same-wave LDS
      ldsP[w][m * 32 + 16 + l16] = (__bf16)p1;   // ordering, no barrier
    }

    // --- O += P * V ; osum += P * ones  (matrix-pipe row sums) ---
    v16bf pf;
    {
      const __bf16* pp = &ldsP[w][0] + l16 * 32;
      v8bf a0 = *(const v8bf*)(pp + lh * 16);
      v8bf a1 = *(const v8bf*)(pp + lh * 16 + 8);
#pragma unroll
      for (int j = 0; j < 8; ++j) { pf[j] = a0[j]; pf[j + 8] = a1[j]; }
    }
    osum = __builtin_amdgcn_wmma_f32_16x16x32_bf16(false, pf, false, onesf,
                                                   (short)0, osum, false,
                                                   false);
#pragma unroll
    for (int nt = 0; nt < 4; ++nt) {
      const __bf16* vp = ldsVt + (nt * 16 + l16) * 32 + lh * 16;
      v8bf b0 = *(const v8bf*)vp;
      v8bf b1 = *(const v8bf*)(vp + 8);
      v16bf vf;
#pragma unroll
      for (int j = 0; j < 8; ++j) { vf[j] = b0[j]; vf[j + 8] = b1[j]; }
      o[nt] = __builtin_amdgcn_wmma_f32_16x16x32_bf16(false, pf, false, vf,
                                                      (short)0, o[nt], false,
                                                      false);
    }
    __syncthreads();   // all ds reads retired before next chunk's staging
  }

  // --- finalize: divide by softmax denominator, write bf16 attn output ---
#pragma unroll
  for (int r = 0; r < 8; ++r) {
    float inv = 1.0f / osum[r];
    int m = qbase + r + lh * 8;
#pragma unroll
    for (int nt = 0; nt < 4; ++nt) {
      O[(size_t)m * C_DIM + h * HD + nt * 16 + l16] = (__bf16)(o[nt][r] * inv);
    }
  }
}

// ---------------------------------------------------------------------------
// Host launch.  Workspace ~144.3 MB:
//   Xbf 26.2M | 4x packed W 3.28M | Q,K,V bf16 3x26.2M | AttO 26.2M | kvrow
// ---------------------------------------------------------------------------
extern "C" void kernel_launch(void* const* d_in, const int* in_sizes, int n_in,
                              void* d_out, int out_size, void* d_ws,
                              size_t ws_size, hipStream_t stream) {
  const float* hs      = (const float*)d_in[0];
  const int*   indices = (const int*)d_in[1];
  const float* Wq      = (const float*)d_in[2];
  const float* Wk      = (const float*)d_in[3];
  const float* Wv      = (const float*)d_in[4];
  const float* Wo      = (const float*)d_in[5];
  const float* bo      = (const float*)d_in[6];
  float*       out     = (float*)d_out;

  char* ws = (char*)d_ws;
  const size_t actBytes = (size_t)M_ROWS * C_DIM * 2;  // 26,214,400
  const size_t wBytes   = (size_t)C_DIM * C_DIM * 2;   //  3,276,800
  __bf16* Xbf  = (__bf16*)ws; ws += actBytes;
  __bf16* Wqp  = (__bf16*)ws; ws += wBytes;
  __bf16* Wkp  = (__bf16*)ws; ws += wBytes;
  __bf16* Wvp  = (__bf16*)ws; ws += wBytes;
  __bf16* Wop  = (__bf16*)ws; ws += wBytes;
  __bf16* Qbf  = (__bf16*)ws; ws += actBytes;
  __bf16* Kbf  = (__bf16*)ws; ws += actBytes;
  __bf16* Vbf  = (__bf16*)ws; ws += actBytes;
  __bf16* AttO = (__bf16*)ws; ws += actBytes;
  int*    kvrow = (int*)ws;

  // 1. convert activations to bf16
  {
    int n8 = M_ROWS * C_DIM / 8;                 // 1,638,400
    cvt_f32_to_bf16<<<(n8 + 255) / 256, 256, 0, stream>>>(hs, Xbf, n8);
  }
  // 2. pack weights into B-fragment layout
  {
    int nT = (C_DIM / 16) * KSTEPS * 32;         // 102,400
    dim3 g((nT + 255) / 256);
    pack_w_bf16<<<g, 256, 0, stream>>>(Wq, Wqp);
    pack_w_bf16<<<g, 256, 0, stream>>>(Wk, Wkp);
    pack_w_bf16<<<g, 256, 0, stream>>>(Wv, Wvp);
    pack_w_bf16<<<g, 256, 0, stream>>>(Wo, Wop);
  }
  // 3. kv gather index table
  {
    int nT = IMGS * 2 * KV_LEN;                  // 30,720
    build_kvrow<<<(nT + 255) / 256, 256, 0, stream>>>(indices, kvrow);
  }
  // 4. Q/K/V projections (bf16 WMMA GEMM, TDM-staged A tiles)
  {
    dim3 grid(C_DIM / 64, M_ROWS / 128);         // (20, 80)
    gemm_wmma<0><<<grid, 128, 0, stream>>>(Xbf, Wqp, Qbf, nullptr, nullptr, nullptr);
    gemm_wmma<0><<<grid, 128, 0, stream>>>(Xbf, Wkp, Kbf, nullptr, nullptr, nullptr);
    gemm_wmma<0><<<grid, 128, 0, stream>>>(Xbf, Wvp, Vbf, nullptr, nullptr, nullptr);
  }
  // 5. flash attention (4 waves share staged K/V per (i, rep, head))
  {
    dim3 grid(T_DIM / 16 / QW, HEADS, IMGS * 2); // (16, 20, 10)
    attn_wmma<<<grid, 128, 0, stream>>>(Qbf, Kbf, Vbf, kvrow, AttO);
  }
  // 6. output projection + bias + residual -> f32 d_out
  {
    dim3 grid(C_DIM / 64, M_ROWS / 128);
    gemm_wmma<1><<<grid, 128, 0, stream>>>(AttO, Wop, nullptr, out, bo, hs);
  }
}
